// utterance_embedding_11845519802536
// MI455X (gfx1250) — compile-verified, matrix operand order checked by reading, not run on hardware
//
#include <hip/hip_runtime.h>
#include <hip/hip_bf16.h>

#define E_DIM 256
#define H_DIM 128
#define G_DIM 384
#define T_DIM 128
#define N_UTT 1024

typedef __attribute__((ext_vector_type(16))) __bf16 v16bf;
typedef __attribute__((ext_vector_type(8)))  float  v8f;
typedef __attribute__((ext_vector_type(8)))  unsigned short us8;
typedef __attribute__((ext_vector_type(4)))  unsigned int v4u;
typedef __attribute__((ext_vector_type(8)))  int v8i_t;
typedef __attribute__((ext_vector_type(4)))  int v4i_t;

union Frag { v16bf bf; us8 h[2]; };

__device__ __forceinline__ unsigned short f2bf(float f) {
  unsigned u = __float_as_uint(f);
  u += 0x7FFFu + ((u >> 16) & 1u);          // round to nearest even
  return (unsigned short)(u >> 16);
}
__device__ __forceinline__ float bf2f(unsigned short h) {
  return __uint_as_float(((unsigned)h) << 16);
}
__device__ __forceinline__ float sigmoidf_(float x) { return 1.f / (1.f + __expf(-x)); }
__device__ __forceinline__ float tanhf_(float x) { float e = __expf(2.f * x); return (e - 1.f) / (e + 1.f); }

// ---------------------------------------------------------------- lens
__global__ void k_lens(const int* __restrict__ d_ids, const int* __restrict__ utt_len,
                       int* __restrict__ lens) {
  int i = blockIdx.x * 256 + threadIdx.x;
  if (i >= N_UTT) return;
  int b = i >> 5, u = i & 31;
  int l = utt_len[d_ids[b] * 32 + u];
  lens[i] = l < 1 ? 1 : (l > T_DIM ? T_DIM : l);
}

// ---------------------------------------------------------------- fp32 -> bf16
__global__ void k_f2bf(const float* __restrict__ src, unsigned short* __restrict__ dst, int n) {
  int i = blockIdx.x * 256 + threadIdx.x;
  if (i < n) dst[i] = f2bf(src[i]);
}

// ---------------------------------------------------------------- embedding gather -> bf16
__global__ void k_gather(const int* __restrict__ x, const float* __restrict__ table,
                         unsigned short* __restrict__ embF) {
  long r = blockIdx.x;                       // 0 .. N*T-1
  int id = x[r];
  const float* src = table + (long)id * E_DIM;
  int tid = threadIdx.x;                     // 64 threads, 4 elems each
  float4 v = *(const float4*)(src + tid * 4);
  unsigned lo = (unsigned)f2bf(v.x) | ((unsigned)f2bf(v.y) << 16);
  unsigned hi = (unsigned)f2bf(v.z) | ((unsigned)f2bf(v.w) << 16);
  uint2 w; w.x = lo; w.y = hi;
  *(uint2*)(embF + r * E_DIM + tid * 4) = w;
}

// ---------------------------------------------------------------- xp = emb @ w_ih^T + b_ih  (bf16 WMMA, f32 accum)
// LDS-staged: per K-chunk of 32, block loads A (64x32) and B (384x32) tiles to LDS,
// then 8 waves (4 rowTiles x 2 colHalves) run 12 WMMA tiles each off ds_load_b128.
__global__ void __launch_bounds__(256, 1)
k_xp_gemm(const unsigned short* __restrict__ A,
          const unsigned short* __restrict__ Wbf,
          const float* __restrict__ bias,
          const int* __restrict__ lens, int rev,
          unsigned short* __restrict__ XP) {
  __shared__ unsigned short sA[64 * 32];     //  4 KB
  __shared__ unsigned short sB[384 * 32];    // 24 KB
  const int tid = threadIdx.x, lane = tid & 31, wave = tid >> 5;
  const int rowTile = wave & 3, colHalf = wave >> 2;
  const int mBase = blockIdx.x * 64;
  const int r15 = lane & 15;
  const int koff = (lane >> 4) * 8;

  // A staging: thread -> (row 0..63, 8-elem k-group); apply utterance-reversal at stage time
  const int aRow = tid >> 2;                 // 0..63
  const int aKg  = (tid & 3) * 8;            // 0,8,16,24
  int srcRow = mBase + aRow;
  if (rev) {                                 // emb_rev[n,t] = emb[n, clip(lens-1-t,0,T-1)]
    int n = srcRow >> 7, t = srcRow & 127;
    int rt = lens[n] - 1 - t;
    rt = rt < 0 ? 0 : rt;
    srcRow = (n << 7) + rt;
  }
  const unsigned short* aSrc = A + (long)srcRow * E_DIM + aKg;

  v8f acc[12] = {};
  for (int k0 = 0; k0 < E_DIM; k0 += 32) {
    __syncthreads();                         // previous chunk fully consumed
    *(us8*)(sA + aRow * 32 + aKg) = *(const us8*)(aSrc + k0);
#pragma unroll
    for (int i = 0; i < 6; ++i) {            // 1536 b128 loads: 384 rows x 4 groups
      const int idx = i * 256 + tid;
      const int bRow = idx >> 2;
      const int bKg = (idx & 3) * 8;
      *(us8*)(sB + bRow * 32 + bKg) = *(const us8*)(Wbf + (long)bRow * E_DIM + k0 + bKg);
    }
    __syncthreads();

    Frag a;
    a.h[0] = *(const us8*)(sA + (rowTile * 16 + r15) * 32 + koff);
    a.h[1] = *(const us8*)(sA + (rowTile * 16 + r15) * 32 + 16 + koff);
#pragma unroll
    for (int i = 0; i < 12; ++i) {
      const int c = colHalf * 12 + i;
      Frag b;
      b.h[0] = *(const us8*)(sB + (c * 16 + r15) * 32 + koff);
      b.h[1] = *(const us8*)(sB + (c * 16 + r15) * 32 + 16 + koff);
      acc[i] = __builtin_amdgcn_wmma_f32_16x16x32_bf16(false, a.bf, false, b.bf,
                                                       (short)0, acc[i], false, false);
    }
  }
  const int hi = lane >> 4;
  const int mRow = mBase + rowTile * 16;
#pragma unroll
  for (int i = 0; i < 12; ++i) {
    const int col = (colHalf * 12 + i) * 16 + r15;
    const float bv = bias[col];
#pragma unroll
    for (int v = 0; v < 8; ++v) {
      XP[(long)(mRow + v + 8 * hi) * G_DIM + col] = f2bf(acc[i][v] + bv);
    }
  }
}

// ---------------------------------------------------------------- GRU recurrence (16 utterances / workgroup)
// w_hh (bf16 [384,128]) DMA'd into LDS via TDM; h-state in LDS (bf16); gh via WMMA; gates in fp32.
__global__ void __launch_bounds__(256, 1)
k_gru(const unsigned short* __restrict__ XP,   // [N,T,384] bf16 (includes b_ih)
      const unsigned short* __restrict__ Whh,  // [384,128] bf16
      const float* __restrict__ bhh,           // [384]
      const int* __restrict__ lens,
      unsigned short* __restrict__ HCAT,       // [N,T,256] bf16
      int backward) {
  extern __shared__ unsigned char smem[];
  unsigned short* whh_lds = (unsigned short*)smem;            // 98304 B
  unsigned short* hbuf    = (unsigned short*)(smem + 98304);  //  4096 B
  float*          gh      = (float*)(smem + 102400);          // 24576 B

  const int tid = threadIdx.x, lane = tid & 31, wave = tid >> 5;
  const int nBase = blockIdx.x * 16;
  const int r15 = lane & 15;
  const int koff = (lane >> 4) * 8;

  if (wave == 0) {
    // Tensor Data Mover: 1-D copy of 49152 bf16 elements (96KB) global -> LDS.
    unsigned long long ga = (unsigned long long)(const void*)Whh;
    unsigned int ldsoff = (unsigned int)(unsigned long long)(void*)whh_lds; // addr[31:0] = LDS offset
    v4u g0;
    g0[0] = 1u;                                            // count=1, user descriptor
    g0[1] = ldsoff;                                        // lds_addr
    g0[2] = (unsigned)(ga & 0xFFFFFFFFull);                // global_addr lo
    g0[3] = (unsigned)((ga >> 32) & 0x1FFFFFFull) | (2u << 30); // addr hi | type=2
    v8i_t g1;
    g1[0] = (int)(1u << 16);        // data_size = 2 bytes
    g1[1] = (int)0xC0000000u;       // tensor_dim0[15:0] = 49152
    g1[2] = 0x00010000;             // tensor_dim1 = 1
    g1[3] = (int)0xC0000000u;       // tile_dim0 = 49152
    g1[4] = 0;                      // tile_dim1/2 unused
    g1[5] = 49152;                  // tensor_dim0_stride lo
    g1[6] = 0; g1[7] = 0;
    v4i_t gz; gz[0] = 0; gz[1] = 0; gz[2] = 0; gz[3] = 0;
    v8i_t gz8; gz8[0] = 0; gz8[1] = 0; gz8[2] = 0; gz8[3] = 0;
    gz8[4] = 0; gz8[5] = 0; gz8[6] = 0; gz8[7] = 0;
    __builtin_amdgcn_tensor_load_to_lds(g0, g1, gz, gz, gz8, 0);
    __builtin_amdgcn_s_wait_tensorcnt(0);
  }
  for (int i = tid; i < 16 * H_DIM; i += 256) hbuf[i] = 0;   // h0 = 0
  __syncthreads();

  for (int s = 0; s < T_DIM; ++s) {
    // gh[16,384] = h[16,128] @ whh^T : 24 col tiles over 8 waves (3 each), K = 128
    v8f acc[3] = {};
#pragma unroll
    for (int k0 = 0; k0 < H_DIM; k0 += 32) {
      Frag a;
      a.h[0] = *(const us8*)(hbuf + r15 * H_DIM + k0 + koff);
      a.h[1] = *(const us8*)(hbuf + r15 * H_DIM + k0 + 16 + koff);
#pragma unroll
      for (int i = 0; i < 3; ++i) {
        const int c = wave * 3 + i;
        const unsigned short* br = whh_lds + (c * 16 + r15) * H_DIM;
        Frag b;
        b.h[0] = *(const us8*)(br + k0 + koff);
        b.h[1] = *(const us8*)(br + k0 + 16 + koff);
        acc[i] = __builtin_amdgcn_wmma_f32_16x16x32_bf16(false, a.bf, false, b.bf,
                                                         (short)0, acc[i], false, false);
      }
    }
    const int hi = lane >> 4;
#pragma unroll
    for (int i = 0; i < 3; ++i) {
      const int col = (wave * 3 + i) * 16 + r15;
      const float bb = bhh[col];
#pragma unroll
      for (int v = 0; v < 8; ++v) gh[(v + 8 * hi) * G_DIM + col] = acc[i][v] + bb;
    }
    __syncthreads();

    // gate math + state update; each thread owns 8 of the 2048 (row,col) elements
#pragma unroll
    for (int e = tid; e < 16 * H_DIM; e += 256) {
      const int row = e >> 7, col = e & 127;
      const int n = nBase + row;
      const unsigned short* xr = XP + ((long)n * T_DIM + s) * G_DIM;
      float r  = sigmoidf_(bf2f(xr[col])             + gh[row * G_DIM + col]);
      float z  = sigmoidf_(bf2f(xr[H_DIM + col])     + gh[row * G_DIM + H_DIM + col]);
      float nn = tanhf_   (bf2f(xr[2 * H_DIM + col]) + r * gh[row * G_DIM + 2 * H_DIM + col]);
      float hold = bf2f(hbuf[row * H_DIM + col]);
      float hnew = (1.f - z) * nn + z * hold;
      const int l = lens[n];
      if (!backward) {
        HCAT[((long)n * T_DIM + s) * E_DIM + col] = f2bf(hnew);
      } else if (s < l) {
        HCAT[((long)n * T_DIM + (l - 1 - s)) * E_DIM + H_DIM + col] = f2bf(hnew);
      }
      hbuf[row * H_DIM + col] = f2bf(hnew);
    }
    __syncthreads();
  }
}

// ---------------------------------------------------------------- fused attention pooling + output head
// score_t = ((wk^T q)·h_t + q·bk)/16 ; pooled = Σ attn_t h_t ; out = wl(wo(wv pooled + bv)+bo)+bl
__global__ void k_attn(const unsigned short* __restrict__ HCAT, const int* __restrict__ lens,
                       const float* __restrict__ wq, const float* __restrict__ bq,
                       const float* __restrict__ wk, const float* __restrict__ bk,
                       const float* __restrict__ wv, const float* __restrict__ bv,
                       const float* __restrict__ wo, const float* __restrict__ bo,
                       const float* __restrict__ wl, const float* __restrict__ bl,
                       float* __restrict__ out) {
  __shared__ float hlast[E_DIM], q[E_DIM], qk[E_DIM], vecA[E_DIM], vecB[E_DIM], red[256];
  __shared__ float sc[T_DIM];
  const int n = blockIdx.x, tid = threadIdx.x;
  const int l = lens[n];
  const unsigned short* hn = HCAT + (long)n * T_DIM * E_DIM;

  hlast[tid] = bf2f(hn[(long)(l - 1) * E_DIM + tid]);
  __syncthreads();
  {
    const float* wr = wq + (long)tid * E_DIM;
    float s = 0.f;
    for (int j = 0; j < E_DIM; ++j) s += wr[j] * hlast[j];
    q[tid] = s + bq[tid];
  }
  __syncthreads();
  {
    float s = 0.f;
    for (int j = 0; j < E_DIM; ++j) s += wk[(long)j * E_DIM + tid] * q[j];
    qk[tid] = s;
    red[tid] = q[tid] * bk[tid];
  }
  __syncthreads();
  for (int o = 128; o > 0; o >>= 1) { if (tid < o) red[tid] += red[tid + o]; __syncthreads(); }
  const float c0 = red[0];

  if (tid < T_DIM) {
    float sv = -3.0e38f;
    if (tid < l) {
      const unsigned short* hr = hn + (long)tid * E_DIM;
      float d = 0.f;
      for (int j = 0; j < E_DIM; ++j) d += qk[j] * bf2f(hr[j]);
      sv = (d + c0) * 0.0625f;                 // /sqrt(256)
    }
    sc[tid] = sv;
  }
  __syncthreads();
  red[tid] = (tid < T_DIM) ? sc[tid] : -3.0e38f;
  __syncthreads();
  for (int o = 128; o > 0; o >>= 1) { if (tid < o) red[tid] = fmaxf(red[tid], red[tid + o]); __syncthreads(); }
  const float mx = red[0];
  __syncthreads();
  if (tid < T_DIM) sc[tid] = (tid < l) ? __expf(sc[tid] - mx) : 0.f;
  __syncthreads();
  red[tid] = (tid < T_DIM) ? sc[tid] : 0.f;
  __syncthreads();
  for (int o = 128; o > 0; o >>= 1) { if (tid < o) red[tid] += red[tid + o]; __syncthreads(); }
  const float inv = 1.f / red[0];

  {
    float p = 0.f;
    for (int t = 0; t < l; ++t) p += sc[t] * bf2f(hn[(long)t * E_DIM + tid]);
    vecA[tid] = p * inv;                       // pooled
  }
  __syncthreads();
  { const float* wr = wv + (long)tid * E_DIM; float s = 0.f;
    for (int j = 0; j < E_DIM; ++j) s += wr[j] * vecA[j];
    vecB[tid] = s + bv[tid]; }                 // ctx
  __syncthreads();
  { const float* wr = wo + (long)tid * E_DIM; float s = 0.f;
    for (int j = 0; j < E_DIM; ++j) s += wr[j] * vecB[j];
    vecA[tid] = s + bo[tid]; }                 // make_output
  __syncthreads();
  { const float* wr = wl + (long)tid * E_DIM; float s = 0.f;
    for (int j = 0; j < E_DIM; ++j) s += wr[j] * vecA[j];
    out[(long)n * E_DIM + tid] = s + bl[tid]; }
}

// ---------------------------------------------------------------- launch
extern "C" void kernel_launch(void* const* d_in, const int* in_sizes, int n_in,
                              void* d_out, int out_size, void* d_ws, size_t ws_size,
                              hipStream_t stream) {
  (void)in_sizes; (void)n_in; (void)out_size; (void)ws_size;
  const int*   x       = (const int*)d_in[0];
  const int*   d_ids   = (const int*)d_in[1];
  const int*   utt_len = (const int*)d_in[2];
  const float* emb_tab = (const float*)d_in[3];
  const float* w_ih_f  = (const float*)d_in[4];
  const float* w_hh_f  = (const float*)d_in[5];
  const float* b_ih_f  = (const float*)d_in[6];
  const float* b_hh_f  = (const float*)d_in[7];
  const float* w_ih_b  = (const float*)d_in[8];
  const float* w_hh_b  = (const float*)d_in[9];
  const float* b_ih_b  = (const float*)d_in[10];
  const float* b_hh_b  = (const float*)d_in[11];
  const float* wq = (const float*)d_in[12]; const float* bq = (const float*)d_in[13];
  const float* wk = (const float*)d_in[14]; const float* bk = (const float*)d_in[15];
  const float* wv = (const float*)d_in[16]; const float* bv = (const float*)d_in[17];
  const float* wo = (const float*)d_in[18]; const float* bo = (const float*)d_in[19];
  const float* wl = (const float*)d_in[20]; const float* bl = (const float*)d_in[21];

  unsigned char* ws = (unsigned char*)d_ws;
  size_t off = 0;
  auto take = [&](size_t bytes) -> unsigned char* {
    unsigned char* p = ws + off;
    off = (off + bytes + 255) & ~(size_t)255;
    return p;
  };
  int*            lens = (int*)take((size_t)N_UTT * 4);
  unsigned short* wihF = (unsigned short*)take((size_t)G_DIM * E_DIM * 2);
  unsigned short* whhF = (unsigned short*)take((size_t)G_DIM * H_DIM * 2);
  unsigned short* wihB = (unsigned short*)take((size_t)G_DIM * E_DIM * 2);
  unsigned short* whhB = (unsigned short*)take((size_t)G_DIM * H_DIM * 2);
  unsigned short* embF = (unsigned short*)take((size_t)N_UTT * T_DIM * E_DIM * 2);
  unsigned short* xp   = (unsigned short*)take((size_t)N_UTT * T_DIM * G_DIM * 2);
  unsigned short* hcat = (unsigned short*)take((size_t)N_UTT * T_DIM * E_DIM * 2);

  const int gruLds = 98304 + 4096 + 24576;   // 126976 B
  (void)hipFuncSetAttribute(reinterpret_cast<const void*>(&k_gru),
                            hipFuncAttributeMaxDynamicSharedMemorySize, gruLds);

  k_lens<<<4, 256, 0, stream>>>(d_ids, utt_len, lens);
  k_f2bf<<<(G_DIM * E_DIM + 255) / 256, 256, 0, stream>>>(w_ih_f, wihF, G_DIM * E_DIM);
  k_f2bf<<<(G_DIM * H_DIM + 255) / 256, 256, 0, stream>>>(w_hh_f, whhF, G_DIM * H_DIM);
  k_f2bf<<<(G_DIM * E_DIM + 255) / 256, 256, 0, stream>>>(w_ih_b, wihB, G_DIM * E_DIM);
  k_f2bf<<<(G_DIM * H_DIM + 255) / 256, 256, 0, stream>>>(w_hh_b, whhB, G_DIM * H_DIM);
  k_gather<<<N_UTT * T_DIM, 64, 0, stream>>>(x, emb_tab, embF);

  k_xp_gemm<<<N_UTT * T_DIM / 64, 256, 0, stream>>>(embF, wihF, b_ih_f, lens, 0, xp);
  k_gru<<<N_UTT / 16, 256, gruLds, stream>>>(xp, whhF, b_hh_f, lens, hcat, 0);
  k_xp_gemm<<<N_UTT * T_DIM / 64, 256, 0, stream>>>(embF, wihB, b_ih_b, lens, 1, xp);
  k_gru<<<N_UTT / 16, 256, gruLds, stream>>>(xp, whhB, b_hh_b, lens, hcat, 1);

  k_attn<<<N_UTT, 256, 0, stream>>>(hcat, lens, wq, bq, wk, bk, wv, bv, wo, bo, wl, bl,
                                    (float*)d_out);
}
